// LAHRv4_40192303956224
// MI455X (gfx1250) — matrix operand order repaired and checked by reading, not compiled, not run on hardware
//
#include <hip/hip_runtime.h>
#include <math.h>

// ---------------- model constants ----------------
#define D_     1024
#define TD3    3072
#define H_     16
#define HD_    64
#define T_     1024
#define B_     2
#define DFF_   2730
#define NMEM   1024
#define KMEM   8
#define KMOD   128
#define NVT    4
#define LW_    1028          // N_VT + T
#define VOCAB_ 50257

// ---------------- WMMA types ----------------
typedef __bf16 bf16_t;
typedef bf16_t v16bf __attribute__((ext_vector_type(16)));
typedef float  v8f   __attribute__((ext_vector_type(8)));

struct __align__(16) U128 { unsigned int x, y, z, w; };
union FragU { U128 u[2]; v16bf v; };

__device__ __forceinline__ v8f vzero8() {
  v8f v;
#pragma unroll
  for (int i = 0; i < 8; ++i) v[i] = 0.f;
  return v;
}

__device__ __forceinline__ unsigned short f2bf(float f) {
  unsigned int u = __float_as_uint(f);
  unsigned int r = u + 0x7FFFu + ((u >> 16) & 1u);  // round-to-nearest-even
  return (unsigned short)(r >> 16);
}
__device__ __forceinline__ unsigned pack2bf(float a, float b) {
  return (unsigned)f2bf(a) | ((unsigned)f2bf(b) << 16);
}

__device__ __forceinline__ v8f wmma_bf16(v16bf a, v16bf b, v8f c) {
  return __builtin_amdgcn_wmma_f32_16x16x32_bf16(false, a, false, b, (short)0, c, false, false);
}

// async global->LDS copy of 16 bytes per lane (CDNA5, tracked by ASYNCcnt).
// Generic shared-pointer low 32 bits == LDS byte offset (ISA 10.2 aperture rules).
__device__ __forceinline__ void async_b128(void* lds, const void* gaddr) {
  unsigned l = (unsigned)(size_t)lds;
  asm volatile("global_load_async_to_lds_b128 %0, %1, off"
               :: "v"(l), "v"(gaddr) : "memory");
}
__device__ __forceinline__ void async_wait0() {
  asm volatile("s_wait_asynccnt 0" ::: "memory");
}

// ---------------- generic GEMM: C[M,N] = A[M,K] @ W[N,K]^T ----------------
// 128 threads (4 waves), 64x64 tile, BK=32, bf16 WMMA with f32 accum.
// Fast path: async global->LDS (f32), then LDS-side bf16 packing. Edge tiles:
// branchless clamped loads.
#define LSTR 48
__global__ __launch_bounds__(128) void k_gemm(const float* __restrict__ A,
                                              const float* __restrict__ W,
                                              float* __restrict__ C,
                                              int M, int N, int K) {
  __shared__ __align__(16) float A32[64 * 32];
  __shared__ __align__(16) float W32[64 * 32];
  __shared__ __align__(16) unsigned short As[64 * LSTR];
  __shared__ __align__(16) unsigned short Ws[64 * LSTR];
  const int tid = threadIdx.x;
  const int lane = tid & 31, wv = tid >> 5;
  const int g = lane >> 4, ln16 = lane & 15;
  const int m0 = blockIdx.y * 64, n0 = blockIdx.x * 64;
  const bool tileFull = (m0 + 64 <= M) && (n0 + 64 <= N);
  v8f acc[4];
#pragma unroll
  for (int t = 0; t < 4; ++t) acc[t] = vzero8();
  const int ktn = (K + 31) >> 5;
  for (int kt = 0; kt < ktn; ++kt) {
    const int k0 = kt << 5;
    const bool fast = tileFull && (k0 + 32 <= K);
    if (fast) {
#pragma unroll
      for (int i = 0; i < 4; ++i) {
        int q = i * 128 + tid;            // 512 16B chunks per tile
        int r = q >> 3, c4 = (q & 7) << 2;
        async_b128(&A32[r * 32 + c4], &A[(size_t)(m0 + r) * K + k0 + c4]);
        async_b128(&W32[r * 32 + c4], &W[(size_t)(n0 + r) * K + k0 + c4]);
      }
      async_wait0();
    }
    __syncthreads();   // prev WMMA reads done; async LDS data visible to all waves
    if (fast) {
#pragma unroll
      for (int i = 0; i < 8; ++i) {
        int p = i * 128 + tid;            // 1024 bf16 pairs
        int r = p >> 4, c2 = (p & 15) << 1;
        *(unsigned*)&As[r * LSTR + c2] = pack2bf(A32[p * 2], A32[p * 2 + 1]);
        *(unsigned*)&Ws[r * LSTR + c2] = pack2bf(W32[p * 2], W32[p * 2 + 1]);
      }
    } else {
#pragma unroll
      for (int i = 0; i < 8; ++i) {
        int p = i * 128 + tid;
        int r = p >> 4, c2 = (p & 15) << 1;
        int mr = m0 + r, nr = n0 + r;
        int kc0 = k0 + c2, kc1 = kc0 + 1;
        int mra = min(mr, M - 1), nra = min(nr, N - 1);
        int k0a = min(kc0, K - 1), k1a = min(kc1, K - 1);
        float a0 = A[(size_t)mra * K + k0a]; a0 = (mr < M && kc0 < K) ? a0 : 0.f;
        float a1 = A[(size_t)mra * K + k1a]; a1 = (mr < M && kc1 < K) ? a1 : 0.f;
        float w0 = W[(size_t)nra * K + k0a]; w0 = (nr < N && kc0 < K) ? w0 : 0.f;
        float w1 = W[(size_t)nra * K + k1a]; w1 = (nr < N && kc1 < K) ? w1 : 0.f;
        *(unsigned*)&As[r * LSTR + c2] = pack2bf(a0, a1);
        *(unsigned*)&Ws[r * LSTR + c2] = pack2bf(w0, w1);
      }
    }
    __syncthreads();
    FragU fa;
    const int am = wv * 16 + ln16;
    fa.u[0] = *(const U128*)&As[am * LSTR + g * 16];
    fa.u[1] = *(const U128*)&As[am * LSTR + g * 16 + 8];
    FragU fb[4];
#pragma unroll
    for (int nt = 0; nt < 4; ++nt) {
      const unsigned short* wr = &Ws[(nt * 16 + ln16) * LSTR + g * 16];
      fb[nt].u[0] = *(const U128*)wr;
      fb[nt].u[1] = *(const U128*)(wr + 8);
    }
#pragma unroll
    for (int nt = 0; nt < 4; ++nt)
      acc[nt] = wmma_bf16(fa.v, fb[nt].v, acc[nt]);
  }
#pragma unroll
  for (int nt = 0; nt < 4; ++nt) {
#pragma unroll
    for (int r = 0; r < 8; ++r) {
      int m = m0 + wv * 16 + g * 8 + r;
      int n = n0 + nt * 16 + ln16;
      if (m < M && n < N) C[(size_t)m * N + n] = acc[nt][r];
    }
  }
}

// ---------------- flash attention: per (qtile64, head, batch) ----------------
// qkv rows: [q(1024) | k(1024) | v(1024)], head slice h*64. Streaming softmax,
// QK^T and P@V both via bf16 WMMA. pos==nullptr -> identity positions.
__global__ __launch_bounds__(128) void k_attn(const float* __restrict__ qkv,
                                              float* __restrict__ out,
                                              const int* __restrict__ pos,
                                              int L) {
  __shared__ __align__(16) unsigned short Qs[64 * 80];
  __shared__ __align__(16) unsigned short Ks[32 * 80];
  __shared__ __align__(16) unsigned short Vt[64 * 48];
  __shared__ __align__(16) unsigned short Pb[64 * 48];
  __shared__ float Sf[64 * 36];
  __shared__ float Mrow[64], Lrow[64], Crow[64];
  const int tid = threadIdx.x;
  const int lane = tid & 31, wv = tid >> 5;
  const int g = lane >> 4, ln16 = lane & 15;
  const int qt = blockIdx.x, h = blockIdx.y, b = blockIdx.z;
  const size_t base = (size_t)b * L;
#pragma unroll
  for (int i = 0; i < 16; ++i) {       // stage Q 64x64 (bf16 pairs)
    int p = i * 128 + tid;             // 2048 pairs
    int r = p >> 5, c2 = (p & 31) << 1;
    int qi = qt * 64 + r, qa = min(qi, L - 1);
    float v0 = qkv[(base + qa) * TD3 + h * HD_ + c2];
    float v1 = qkv[(base + qa) * TD3 + h * HD_ + c2 + 1];
    v0 = (qi < L) ? v0 : 0.f;
    v1 = (qi < L) ? v1 : 0.f;
    *(unsigned*)&Qs[r * 80 + c2] = pack2bf(v0, v1);
  }
  if (tid < 64) { Mrow[tid] = -3.0e38f; Lrow[tid] = 0.f; }
  v8f o[4];
#pragma unroll
  for (int t = 0; t < 4; ++t) o[t] = vzero8();
  const int nkt = (L + 31) >> 5;
  for (int kt = 0; kt < nkt; ++kt) {
#pragma unroll
    for (int i = 0; i < 16; ++i) {     // stage K 32x64, V transposed
      int e = i * 128 + tid;
      int r = e >> 6, c = e & 63;
      int kj = kt * 32 + r, ka = min(kj, L - 1);
      float kv = qkv[(base + ka) * TD3 + D_ + h * HD_ + c];
      float vv = qkv[(base + ka) * TD3 + 2 * D_ + h * HD_ + c];
      kv = (kj < L) ? kv : 0.f;
      vv = (kj < L) ? vv : 0.f;
      Ks[r * 80 + c] = f2bf(kv);
      Vt[c * 48 + r] = f2bf(vv);
    }
    __syncthreads();
    {                                  // S = Q @ K^T  (64x32)
      v8f s0 = vzero8(), s1 = vzero8();
      const int am = wv * 16 + ln16;
#pragma unroll
      for (int kk = 0; kk < 2; ++kk) {
        const int kk0 = kk * 32;
        FragU fa, fb0, fb1;
        fa.u[0]  = *(const U128*)&Qs[am * 80 + kk0 + g * 16];
        fa.u[1]  = *(const U128*)&Qs[am * 80 + kk0 + g * 16 + 8];
        fb0.u[0] = *(const U128*)&Ks[ln16 * 80 + kk0 + g * 16];
        fb0.u[1] = *(const U128*)&Ks[ln16 * 80 + kk0 + g * 16 + 8];
        fb1.u[0] = *(const U128*)&Ks[(16 + ln16) * 80 + kk0 + g * 16];
        fb1.u[1] = *(const U128*)&Ks[(16 + ln16) * 80 + kk0 + g * 16 + 8];
        s0 = wmma_bf16(fa.v, fb0.v, s0);
        s1 = wmma_bf16(fa.v, fb1.v, s1);
      }
#pragma unroll
      for (int r = 0; r < 8; ++r) {
        int row = wv * 16 + g * 8 + r;
        Sf[row * 36 + ln16] = s0[r];
        Sf[row * 36 + 16 + ln16] = s1[r];
      }
    }
    __syncthreads();
    if (tid < 64) {                    // online softmax update
      int row = tid, qi = qt * 64 + row;
      int pq = pos ? pos[b * L + min(qi, L - 1)] : qi;
      float mOld = Mrow[row], mNew = mOld;
      for (int c = 0; c < 32; ++c) {
        int kj = kt * 32 + c;
        int pk = pos ? pos[b * L + min(kj, L - 1)] : kj;
        bool msk = (kj >= L) || (qi >= L) || (pk > pq);
        float sv = Sf[row * 36 + c] * 0.125f;
        if (!msk) mNew = fmaxf(mNew, sv);
      }
      float corr = __expf(mOld - mNew);
      float l = Lrow[row] * corr;
      for (int c = 0; c < 32; ++c) {
        int kj = kt * 32 + c;
        int pk = pos ? pos[b * L + min(kj, L - 1)] : kj;
        bool msk = (kj >= L) || (qi >= L) || (pk > pq);
        float pp = msk ? 0.f : __expf(Sf[row * 36 + c] * 0.125f - mNew);
        l += pp;
        Pb[row * 48 + c] = f2bf(pp);
      }
      Mrow[row] = mNew; Lrow[row] = l; Crow[row] = corr;
    }
    __syncthreads();
    {                                  // O = O*corr + P @ V
      const int am = wv * 16 + ln16;
      FragU fa;
      fa.u[0] = *(const U128*)&Pb[am * 48 + g * 16];
      fa.u[1] = *(const U128*)&Pb[am * 48 + g * 16 + 8];
      FragU fb[4];
#pragma unroll
      for (int nt = 0; nt < 4; ++nt) {
        const unsigned short* vr = &Vt[(nt * 16 + ln16) * 48 + g * 16];
        fb[nt].u[0] = *(const U128*)vr;
        fb[nt].u[1] = *(const U128*)(vr + 8);
      }
#pragma unroll
      for (int nt = 0; nt < 4; ++nt) {
#pragma unroll
        for (int r = 0; r < 8; ++r) o[nt][r] *= Crow[wv * 16 + g * 8 + r];
        o[nt] = wmma_bf16(fa.v, fb[nt].v, o[nt]);
      }
    }
    __syncthreads();
  }
#pragma unroll
  for (int nt = 0; nt < 4; ++nt) {
#pragma unroll
    for (int r = 0; r < 8; ++r) {
      int row = wv * 16 + g * 8 + r;
      int qi = qt * 64 + row;
      if (qi < L) {
        float l = Lrow[row];
        out[(base + qi) * D_ + h * HD_ + nt * 16 + ln16] = (l > 0.f) ? o[nt][r] / l : 0.f;
      }
    }
  }
}

// ---------------- small kernels ----------------
__global__ __launch_bounds__(128) void k_rmsnorm(const float* __restrict__ x,
                                                 const float* __restrict__ w,
                                                 float* __restrict__ o) {
  __shared__ float red[128];
  __shared__ float s_inv;
  const int row = blockIdx.x, tid = threadIdx.x;
  float ss = 0.f;
  for (int c = tid; c < D_; c += 128) { float v = x[(size_t)row * D_ + c]; ss += v * v; }
  red[tid] = ss; __syncthreads();
  for (int s = 64; s > 0; s >>= 1) { if (tid < s) red[tid] += red[tid + s]; __syncthreads(); }
  if (tid == 0) s_inv = rsqrtf(red[0] / (float)D_ + 1e-6f);
  __syncthreads();
  float inv = s_inv;
  for (int c = tid; c < D_; c += 128)
    o[(size_t)row * D_ + c] = x[(size_t)row * D_ + c] * inv * w[c];
}

__global__ void k_embed(const int* __restrict__ ids, const float* __restrict__ emb,
                        const float* __restrict__ pe, float* __restrict__ x) {
  const int row = blockIdx.x, tid = threadIdx.x;
  const int t = row & (T_ - 1);
  const int id = ids[row];
#pragma unroll
  for (int i = 0; i < 4; ++i) {
    int c = i * 256 + tid;
    x[(size_t)row * D_ + c] = emb[(size_t)id * D_ + c] + pe[(size_t)t * D_ + c];
  }
}

__global__ void k_add(float* __restrict__ y, const float* __restrict__ x, int n) {
  int i = blockIdx.x * 256 + threadIdx.x;
  if (i < n) y[i] += x[i];
}
__global__ void k_silu_mul(float* __restrict__ a, const float* __restrict__ b, int n) {
  int i = blockIdx.x * 256 + threadIdx.x;
  if (i < n) { float v = a[i]; a[i] = (v / (1.f + __expf(-v))) * b[i]; }
}
__global__ void k_gelu_bias(float* __restrict__ g, const float* __restrict__ b1, int n, int nc) {
  int i = blockIdx.x * 256 + threadIdx.x;
  if (i < n) { float v = g[i] + b1[i % nc]; g[i] = 0.5f * v * (1.f + erff(v * 0.70710678f)); }
}
__global__ void k_gate_res(float* __restrict__ x, const float* __restrict__ r,
                           const float* __restrict__ s, int n) {
  int i = blockIdx.x * 256 + threadIdx.x;
  if (i < n) { float gt = 1.f / (1.f + __expf(-s[i >> 10])); x[i] += gt * r[i]; }
}
__global__ void k_concat2(float* __restrict__ h, const float* __restrict__ x,
                          const float* __restrict__ r, int n) {
  int i = blockIdx.x * 256 + threadIdx.x;
  if (i < n) {
    int m = i >> 11, c = i & 2047;
    h[i] = (c < D_) ? x[(size_t)m * D_ + c] : r[(size_t)m * D_ + c - D_];
  }
}

__global__ __launch_bounds__(128) void k_rowdot(const float* __restrict__ A,
                                                const float* __restrict__ v,
                                                const float* __restrict__ bias,
                                                float* __restrict__ o, int M, int K) {
  const int lane = threadIdx.x & 31, wv = threadIdx.x >> 5;
  const int row = blockIdx.x * 4 + wv;
  if (row >= M) return;
  float s = 0.f;
  for (int k = lane; k < K; k += 32) s += A[(size_t)row * K + k] * v[k];
#pragma unroll
  for (int m = 16; m > 0; m >>= 1) s += __shfl_xor(s, m, 32);
  if (lane == 0) o[row] = s + (bias ? bias[0] : 0.f);
}

// per-token top-8 over 1024 memory slots + softmax + weighted value gather
__global__ __launch_bounds__(128) void k_mem_topk(const float* __restrict__ sim,
                                                  const float* __restrict__ vals,
                                                  float* __restrict__ rp) {
  __shared__ float sc[NMEM];
  __shared__ float rv[128];
  __shared__ int ri[128];
  __shared__ int sidx[KMEM];
  __shared__ float sw[KMEM];
  const int tok = blockIdx.x, tid = threadIdx.x;
#pragma unroll
  for (int i = 0; i < 8; ++i) sc[i * 128 + tid] = sim[(size_t)tok * NMEM + i * 128 + tid] * 0.03125f;
  __syncthreads();
  for (int j = 0; j < KMEM; ++j) {
    float best = -3.0e38f; int bi = 0;
#pragma unroll
    for (int i = 0; i < 8; ++i) {
      int id = i * 128 + tid;
      float vv = sc[id];
      if (vv > best || (vv == best && id < bi)) { best = vv; bi = id; }
    }
    rv[tid] = best; ri[tid] = bi; __syncthreads();
    for (int s = 64; s > 0; s >>= 1) {
      if (tid < s) {
        if (rv[tid + s] > rv[tid] || (rv[tid + s] == rv[tid] && ri[tid + s] < ri[tid])) {
          rv[tid] = rv[tid + s]; ri[tid] = ri[tid + s];
        }
      }
      __syncthreads();
    }
    if (tid == 0) { sidx[j] = ri[0]; sw[j] = rv[0]; sc[ri[0]] = -3.0e38f; }
    __syncthreads();
  }
  if (tid == 0) {
    float m = sw[0];
    for (int j = 1; j < KMEM; ++j) m = fmaxf(m, sw[j]);
    float sum = 0.f;
    for (int j = 0; j < KMEM; ++j) { sw[j] = __expf(sw[j] - m); sum += sw[j]; }
    for (int j = 0; j < KMEM; ++j) sw[j] /= sum;
  }
  __syncthreads();
#pragma unroll
  for (int i = 0; i < 8; ++i) {
    int col = i * 128 + tid;
    float acc = 0.f;
#pragma unroll
    for (int j = 0; j < KMEM; ++j) acc += sw[j] * vals[(size_t)sidx[j] * D_ + col];
    rp[(size_t)tok * D_ + col] = acc;
  }
}

// per-batch top-128 over 1024 router scores
__global__ __launch_bounds__(256) void k_mod_topk(const float* __restrict__ scores,
                                                  int* __restrict__ idx,
                                                  float* __restrict__ sval) {
  __shared__ float sc[T_];
  __shared__ float rv[256];
  __shared__ int ri[256];
  const int b = blockIdx.x, tid = threadIdx.x;
#pragma unroll
  for (int i = 0; i < 4; ++i) sc[i * 256 + tid] = scores[b * T_ + i * 256 + tid];
  __syncthreads();
  for (int j = 0; j < KMOD; ++j) {
    float best = -3.0e38f; int bi = 0;
#pragma unroll
    for (int i = 0; i < 4; ++i) {
      int id = i * 256 + tid;
      float v = sc[id];
      if (v > best || (v == best && id < bi)) { best = v; bi = id; }
    }
    rv[tid] = best; ri[tid] = bi; __syncthreads();
    for (int s = 128; s > 0; s >>= 1) {
      if (tid < s) {
        if (rv[tid + s] > rv[tid] || (rv[tid + s] == rv[tid] && ri[tid + s] < ri[tid])) {
          rv[tid] = rv[tid + s]; ri[tid] = ri[tid + s];
        }
      }
      __syncthreads();
    }
    if (tid == 0) { idx[b * KMOD + j] = ri[0]; sval[b * KMOD + j] = rv[0]; sc[ri[0]] = -3.0e38f; }
    __syncthreads();
  }
}

__global__ void k_gather(float* __restrict__ dst, const float* __restrict__ x,
                         const int* __restrict__ idx) {
  const int j = blockIdx.x, tid = threadIdx.x;
  const int b = j >> 7;
  const size_t src = (size_t)(b * T_ + idx[j]) * D_;
#pragma unroll
  for (int i = 0; i < 4; ++i) dst[(size_t)j * D_ + i * 256 + tid] = x[src + i * 256 + tid];
}

__global__ void k_scatter(float* __restrict__ x, const float* __restrict__ s0,
                          const float* __restrict__ s1, const int* __restrict__ idx,
                          const float* __restrict__ sval) {
  const int j = blockIdx.x, tid = threadIdx.x;
  const int b = j >> 7;
  const size_t dr = (size_t)(b * T_ + idx[j]) * D_;
  const float gte = 1.f / (1.f + __expf(-sval[j]));
#pragma unroll
  for (int i = 0; i < 4; ++i) {
    int c = i * 256 + tid;
    float a = s0[(size_t)j * D_ + c], pr = s1[(size_t)j * D_ + c];
    x[dr + c] = a + gte * (pr - a);
  }
}

__global__ void k_build_xw(float* __restrict__ xw, const float* __restrict__ x,
                           const float* __restrict__ vt) {
  const int row = blockIdx.x, tid = threadIdx.x;
  const int b = row / LW_, j = row % LW_;
#pragma unroll
  for (int i = 0; i < 4; ++i) {
    int c = i * 256 + tid;
    xw[(size_t)row * D_ + c] =
        (j < NVT) ? vt[(size_t)j * D_ + c] : x[(size_t)(b * T_ + (j - NVT)) * D_ + c];
  }
}

__global__ __launch_bounds__(128) void k_ln_vh(const float* __restrict__ xw,
                                               const float* __restrict__ w,
                                               const float* __restrict__ bsh,
                                               float* __restrict__ o) {
  __shared__ float red[128];
  __shared__ float s_mu, s_inv;
  const int bi = blockIdx.x;
  const int b = bi >> 2, v = bi & 3;
  const size_t row = (size_t)(b * LW_ + v) * D_;
  const int tid = threadIdx.x;
  float s = 0.f;
  for (int c = tid; c < D_; c += 128) s += xw[row + c];
  red[tid] = s; __syncthreads();
  for (int st = 64; st > 0; st >>= 1) { if (tid < st) red[tid] += red[tid + st]; __syncthreads(); }
  if (tid == 0) s_mu = red[0] / (float)D_;
  __syncthreads();
  float mu = s_mu, sv = 0.f;
  for (int c = tid; c < D_; c += 128) { float d = xw[row + c] - mu; sv += d * d; }
  red[tid] = sv; __syncthreads();
  for (int st = 64; st > 0; st >>= 1) { if (tid < st) red[tid] += red[tid + st]; __syncthreads(); }
  if (tid == 0) s_inv = rsqrtf(red[0] / (float)D_ + 1e-5f);
  __syncthreads();
  float inv = s_inv;
  for (int c = tid; c < D_; c += 128)
    o[(size_t)bi * D_ + c] = (xw[row + c] - mu) * inv * w[c] + bsh[c];
}

__global__ void k_write_nv(float* __restrict__ xw, const float* __restrict__ nv,
                           const float* __restrict__ bh) {
  const int bi = blockIdx.x, tid = threadIdx.x;
  const int b = bi >> 2, v = bi & 3;
#pragma unroll
  for (int i = 0; i < 4; ++i) {
    int c = i * 256 + tid;
    xw[(size_t)(b * LW_ + v) * D_ + c] = nv[(size_t)bi * D_ + c] + bh[c];
  }
}

__global__ void k_extract(float* __restrict__ x, const float* __restrict__ xw) {
  const int row = blockIdx.x, tid = threadIdx.x;
  const int b = row >> 10, t = row & (T_ - 1);
#pragma unroll
  for (int i = 0; i < 4; ++i) {
    int c = i * 256 + tid;
    x[(size_t)row * D_ + c] = xw[(size_t)(b * LW_ + NVT + t) * D_ + c];
  }
}

// ---------------- host orchestration ----------------
struct BlockW { const float *n1, *Wqkv, *Wo, *n2, *w1, *w2, *w3; };

static inline void gemm(const float* A, const float* W, float* C, int M, int N, int K,
                        hipStream_t s) {
  dim3 g((N + 63) / 64, (M + 63) / 64);
  k_gemm<<<g, 128, 0, s>>>(A, W, C, M, N, K);
}
static inline int g1d(int n) { return (n + 255) / 256; }

static void run_tblock(float* x, int Lb, int nb, const int* pos, const BlockW& w,
                       float* t1, float* t2, float* t3, float* t4, float* qkv,
                       hipStream_t s) {
  const int M = Lb * nb;
  k_rmsnorm<<<M, 128, 0, s>>>(x, w.n1, t1);
  gemm(t1, w.Wqkv, qkv, M, TD3, D_, s);
  dim3 ag((Lb + 63) / 64, H_, nb);
  k_attn<<<ag, 128, 0, s>>>(qkv, t4, pos, Lb);
  gemm(t4, w.Wo, t1, M, D_, D_, s);
  k_add<<<g1d(M * D_), 256, 0, s>>>(x, t1, M * D_);
  k_rmsnorm<<<M, 128, 0, s>>>(x, w.n2, t1);
  gemm(t1, w.w1, t2, M, DFF_, D_, s);
  gemm(t1, w.w2, t3, M, DFF_, D_, s);
  k_silu_mul<<<g1d(M * DFF_), 256, 0, s>>>(t2, t3, M * DFF_);
  gemm(t2, w.w3, t1, M, D_, DFF_, s);
  k_add<<<g1d(M * D_), 256, 0, s>>>(x, t1, M * D_);
}

extern "C" void kernel_launch(void* const* d_in, const int* in_sizes, int n_in,
                              void* d_out, int out_size, void* d_ws, size_t ws_size,
                              hipStream_t stream) {
  (void)in_sizes; (void)n_in; (void)out_size; (void)ws_size;
  auto F = [&](int i) { return (const float*)d_in[i]; };
  // --- inputs in setup_inputs() dict-insertion order ---
  const int* ids = (const int*)d_in[0];
  const float* embed = F(1);
  const float* pos_embed = F(2);
  const float* mkeys = F(3); const float* mvals = F(4);
  const float* mWq = F(5);   const float* mWo = F(6);
  const float* mg1 = F(7);   const float* mb1 = F(8);
  const float* mg2 = F(9);   const float* mb2 = F(10);
  int p = 11;
  BlockW lyr[6]; const float* rtr[6];
  for (int i = 0; i < 6; ++i) {
    lyr[i] = { F(p), F(p + 1), F(p + 2), F(p + 3), F(p + 4), F(p + 5), F(p + 6) };
    p += 7;
    rtr[i] = nullptr;
    if ((i & 1) == 1) { rtr[i] = F(p); ++p; }
  }
  const float* vt = F(p++);
  BlockW lblk = { F(p), F(p + 1), F(p + 2), F(p + 3), F(p + 4), F(p + 5), F(p + 6) };
  p += 7;
  const float* ln_w = F(p++); const float* ln_b = F(p++);
  const float* W_h2e = F(p++); const float* b_h2e = F(p++);
  const float* fnorm = F(p++);

  // --- workspace partition ---
  char* ws = (char*)d_ws;
  size_t off = 0;
  auto alloc = [&](size_t bytes) -> char* {
    char* r = ws + off;
    off = (off + bytes + 255) & ~(size_t)255;
    return r;
  };
  const int MT = B_ * T_;     // 2048
  const int MW = B_ * LW_;    // 2056
  float* X    = (float*)alloc((size_t)MT * D_ * 4);
  float* XW   = (float*)alloc((size_t)MW * D_ * 4);
  float* QKV  = (float*)alloc((size_t)MW * TD3 * 4);   // also reused as concat buffer
  float* T1   = (float*)alloc((size_t)MW * D_ * 4);
  float* T2   = (float*)alloc((size_t)MW * DFF_ * 4);
  float* T3   = (float*)alloc((size_t)MW * DFF_ * 4);
  float* T4   = (float*)alloc((size_t)MW * D_ * 4);
  float* SEL0 = (float*)alloc((size_t)B_ * KMOD * D_ * 4);
  float* SEL1 = (float*)alloc((size_t)B_ * KMOD * D_ * 4);
  float* SCORES = (float*)alloc((size_t)MT * 4);
  float* SVAL = (float*)alloc((size_t)B_ * KMOD * 4);
  int*   IDXM = (int*)  alloc((size_t)B_ * KMOD * 4);
  float* GATE = (float*)alloc((size_t)MT * 4);
  float* VT8  = (float*)alloc((size_t)B_ * NVT * D_ * 4);

  // 1) embedding
  k_embed<<<MT, 256, 0, stream>>>(ids, embed, pos_embed, X);
  // 2) product-key memory
  gemm(X, mWq, T1, MT, D_, D_, stream);                 // q
  gemm(T1, mkeys, T2, MT, NMEM, D_, stream);            // sim (unscaled)
  k_mem_topk<<<MT, 128, 0, stream>>>(T2, mvals, T3);    // retrieved_pre
  gemm(T3, mWo, T4, MT, D_, D_, stream);                // retrieved
  k_concat2<<<g1d(MT * 2 * D_), 256, 0, stream>>>(QKV, X, T4, MT * 2 * D_);
  gemm(QKV, mg1, T1, MT, D_ / 2, 2 * D_, stream);
  k_gelu_bias<<<g1d(MT * D_ / 2), 256, 0, stream>>>(T1, mb1, MT * D_ / 2, D_ / 2);
  k_rowdot<<<(MT + 3) / 4, 128, 0, stream>>>(T1, mg2, mb2, GATE, MT, D_ / 2);
  k_gate_res<<<g1d(MT * D_), 256, 0, stream>>>(X, T4, GATE, MT * D_);
  // 3) transformer / mixture-of-depth layers
  for (int i = 0; i < 6; ++i) {
    if (!rtr[i]) {
      run_tblock(X, T_, B_, nullptr, lyr[i], T1, T2, T3, T4, QKV, stream);
    } else {
      k_rowdot<<<(MT + 3) / 4, 128, 0, stream>>>(X, rtr[i], nullptr, SCORES, MT, D_);
      k_mod_topk<<<B_, 256, 0, stream>>>(SCORES, IDXM, SVAL);
      k_gather<<<B_ * KMOD, 256, 0, stream>>>(SEL0, X, IDXM);
      k_gather<<<B_ * KMOD, 256, 0, stream>>>(SEL1, X, IDXM);
      run_tblock(SEL1, KMOD, B_, IDXM, lyr[i], T1, T2, T3, T4, QKV, stream);
      k_scatter<<<B_ * KMOD, 256, 0, stream>>>(X, SEL0, SEL1, IDXM, SVAL);
    }
  }
  // 4) latent reasoning
  k_build_xw<<<MW, 256, 0, stream>>>(XW, X, vt);
  for (int it = 0; it < 4; ++it) {
    run_tblock(XW, LW_, B_, nullptr, lblk, T1, T2, T3, T4, QKV, stream);
    k_ln_vh<<<B_ * NVT, 128, 0, stream>>>(XW, ln_w, ln_b, VT8);
    gemm(VT8, W_h2e, T1, B_ * NVT, D_, D_, stream);
    k_write_nv<<<B_ * NVT, 256, 0, stream>>>(XW, T1, b_h2e);
  }
  k_extract<<<MT, 256, 0, stream>>>(X, XW);
  // 5) final norm + logits
  k_rmsnorm<<<MT, 128, 0, stream>>>(X, fnorm, T1);
  gemm(T1, embed, (float*)d_out, MT, VOCAB_, D_, stream);
}
